// SSP_Net_58712202936967
// MI455X (gfx1250) — compile-verified, matrix operand order checked by reading, not run on hardware
//
#include <hip/hip_runtime.h>

typedef __attribute__((ext_vector_type(16))) _Float16 v16h;
typedef __attribute__((ext_vector_type(8)))  float    v8f;

#define B_ROWS 65536
#define Q_DIM  64
#define L_DIM  512
#define T_ITER 16

// ---- workspace layout (in halves for the packed f16 region) ----
// S packed:      512x512 halves   (16 ktiles x 32 ntiles, 512 halves/block)
// Dict packed:   64x512  halves   ( 2 ktiles x 32 ntiles)
// Dict^T packed: 512x64  halves   (16 ktiles x  4 ntiles)
#define SPACK_OFF   0
#define DPACK_OFF   (512*512)
#define DTPACK_OFF  (512*512 + 64*512)
#define WS_HALVES   (512*512 + 64*512 + 512*64)      // 640 blocks * 512
#define LTHR_BYTE_OFF ((size_t)WS_HALVES * 2)        // then B floats of lam/c

// Per-lane K ordering of a 16-bit WMMA fragment (ISA 7.12.2):
// lanes 0-15: K = kb+0..7 then 16+kb..23 ; lanes 16-31: kb=8 -> K 8..15, 24..31
__device__ __forceinline__ int kSeq(int lane, int j) {
  int kb = (lane >> 4) << 3;
  return (j < 8) ? (kb + j) : (16 + kb + (j - 8));
}

// ================= kernel 1: pack S / Dict / Dict^T into f16 fragments ======
__global__ void pack_kernel(const float* __restrict__ Dict,
                            const float* __restrict__ cptr,
                            _Float16* __restrict__ wsh) {
  int tid = blockIdx.x * blockDim.x + threadIdx.x;
  if (tid >= WS_HALVES) return;
  int blk  = tid >> 9;          // 512 halves per fragment block
  int pos  = tid & 511;
  int lane = pos >> 4;
  int j    = pos & 15;
  int nl   = lane & 15;
  int kl   = kSeq(lane, j);
  float v;
  if (blk < 512) {              // S = I - (D^T D)/c  (symmetric, so .T is free)
    int kt = blk >> 5, nt = blk & 31;
    int k = kt * 32 + kl, n = nt * 16 + nl;
    float inv_c = 1.0f / cptr[0];
    float dot = 0.0f;
    #pragma unroll 8
    for (int q = 0; q < 64; ++q) dot += Dict[q * 512 + k] * Dict[q * 512 + n];
    v = ((k == n) ? 1.0f : 0.0f) - dot * inv_c;
  } else if (blk < 576) {       // Dict (Q x L) as B-matrix for y = x @ Dict
    int b2 = blk - 512;
    int kt = b2 >> 5, nt = b2 & 31;
    int k = kt * 32 + kl, n = nt * 16 + nl;
    v = Dict[k * 512 + n];
  } else {                      // Dict^T (L x Q) as B-matrix for x_pred
    int b3 = blk - 576;
    int kt = b3 >> 2, nt = b3 & 3;
    int k = kt * 32 + kl, n = nt * 16 + nl;
    v = Dict[n * 512 + k];
  }
  wsh[tid] = (_Float16)v;
}

// ================= kernel 2: lambda-predictor MLP (thread per row) ==========
__global__ void mlp_kernel(const float* __restrict__ x,
                           const float* __restrict__ W1, const float* __restrict__ b1,
                           const float* __restrict__ W2, const float* __restrict__ b2,
                           const float* __restrict__ W3, const float* __restrict__ b3,
                           const float* __restrict__ W4, const float* __restrict__ b4,
                           const float* __restrict__ cptr,
                           float* __restrict__ lam_out, float* __restrict__ lthr) {
  extern __shared__ float sm[];
  float* sW1 = sm;              // 8192
  float* sW2 = sm + 8192;       // 8192
  float* sW3 = sm + 16384;      // 2048
  float* sW4 = sm + 18432;      // 32
  float* sb1 = sm + 18464;      // 128
  float* sb2 = sm + 18592;      // 64
  float* sb3 = sm + 18656;      // 32
  float* sb4 = sm + 18688;      // 1
  float* sh1 = sm + 18720;      // 128 rows * 128 threads (h1 staging)

  int tid = threadIdx.x;        // blockDim = 128
  for (int i = tid; i < 8192; i += 128) sW1[i] = W1[i];
  for (int i = tid; i < 8192; i += 128) sW2[i] = W2[i];
  for (int i = tid; i < 2048; i += 128) sW3[i] = W3[i];
  if (tid < 32) sW4[tid] = W4[tid];
  sb1[tid] = b1[tid];
  if (tid < 64) sb2[tid] = b2[tid];
  if (tid < 32) sb3[tid] = b3[tid];
  if (tid == 0) sb4[0] = b4[0];
  __syncthreads();

  int row = blockIdx.x * 128 + tid;
  float xr[64];
  #pragma unroll 4
  for (int q = 0; q < 64; ++q) xr[q] = x[row * 64 + q];

  for (int jo = 0; jo < 128; ++jo) {
    float s = sb1[jo];
    #pragma unroll 8
    for (int q = 0; q < 64; ++q) s += sW1[jo * 64 + q] * xr[q];
    sh1[jo * 128 + tid] = fmaxf(s, 0.0f);   // stride-128: conflict-free per wave
  }
  float h2[64];
  for (int jo = 0; jo < 64; ++jo) {
    float s = sb2[jo];
    #pragma unroll 8
    for (int q = 0; q < 128; ++q) s += sW2[jo * 128 + q] * sh1[q * 128 + tid];
    h2[jo] = fmaxf(s, 0.0f);
  }
  float h3[32];
  for (int jo = 0; jo < 32; ++jo) {
    float s = sb3[jo];
    #pragma unroll 8
    for (int q = 0; q < 64; ++q) s += sW3[jo * 64 + q] * h2[q];
    h3[jo] = fmaxf(s, 0.0f);
  }
  float lam = sb4[0] + 2.0f;
  #pragma unroll 8
  for (int q = 0; q < 32; ++q) lam += sW4[q] * h3[q];
  lam_out[row] = lam;
  lthr[row] = lam / cptr[0];
}

// ================= kernel 3: block-cooperative WMMA ISTA ====================
__device__ __forceinline__ v16h load_a_frag(const _Float16* base, int row_stride,
                                            int kc, int lane) {
  int m  = lane & 15;
  int kb = kc * 32 + ((lane >> 4) << 3);
  const _Float16* p = base + m * row_stride + kb;
  union { uint4 u[2]; v16h v; } r;
  r.u[0] = *(const uint4*)(p);        // K kb..kb+7      -> ds_load_b128
  r.u[1] = *(const uint4*)(p + 16);   // K 16+kb..16+kb+7
  return r.v;
}
__device__ __forceinline__ v16h load_b_frag(const _Float16* blkbase, int lane) {
  const uint4* p = (const uint4*)(blkbase + lane * 16);  // pre-swizzled by pack
  union { uint4 u[2]; v16h v; } r;
  r.u[0] = p[0]; r.u[1] = p[1];       // two global_load_b128
  return r.v;
}
__device__ __forceinline__ float soft_thr(float v, float l) {
  float s = fabsf(v) - l;
  s = s > 0.0f ? s : 0.0f;
  return copysignf(s, v);
}

// Block geometry: 64 rows (4 strips of 16), 8 waves, each wave owns 4 n-tiles.
// Per K-step a wave issues 8 WMMAs (4 strips x 2 n-chains): the dependent
// distance of 8 fully covers the f16 WMMA->WMMA RAW hazard (5 slots).
#define BLOCK_M   64
#define THREADS   256
// LDS (block-shared): alphaA 64K | alphaB 64K | yfrag 128K | sthr 256B
#define ALPHA_A_OFF 0
#define ALPHA_B_OFF 65536
#define YFRAG_OFF   131072
#define STHR_OFF    262144
#define LDS_BYTES   262400

template <bool LAST>
__device__ __forceinline__ void ista_iter(const _Float16* __restrict__ cur,
                                          _Float16* __restrict__ nxt,
                                          const _Float16* __restrict__ Spack,
                                          const float* __restrict__ yfrag,
                                          const float* __restrict__ sthr,
                                          float* __restrict__ out_alpha,
                                          int rowBase, int lane, int hi,
                                          int ntBase) {
  int nl = lane & 15;
  for (int np = 0; np < 2; ++np) {
    int nt0 = ntBase + np * 2, nt1 = nt0 + 1;
    v8f acc0[4] = {}, acc1[4] = {};
    #pragma unroll 4
    for (int kc = 0; kc < 16; ++kc) {
      v16h b0 = load_b_frag(Spack + (kc * 32 + nt0) * 512, lane);
      v16h b1 = load_b_frag(Spack + (kc * 32 + nt1) * 512, lane);
      if (kc < 15) __builtin_prefetch(Spack + ((kc + 1) * 32 + nt0) * 512, 0, 1);
      #pragma unroll
      for (int s = 0; s < 4; ++s) {
        v16h a = load_a_frag(cur + s * 16 * 512, 512, kc, lane);
        acc0[s] = __builtin_amdgcn_wmma_f32_16x16x32_f16(false, a, false, b0,
                                                         (short)0, acc0[s],
                                                         false, false);
        acc1[s] = __builtin_amdgcn_wmma_f32_16x16x32_f16(false, a, false, b1,
                                                         (short)0, acc1[s],
                                                         false, false);
      }
    }
    #pragma unroll
    for (int s = 0; s < 4; ++s) {
      const float4* tp = (const float4*)(sthr + s * 16 + hi * 8);
      float4 t0 = tp[0], t1 = tp[1];
      float thr[8] = {t0.x, t0.y, t0.z, t0.w, t1.x, t1.y, t1.z, t1.w};
      #pragma unroll
      for (int half = 0; half < 2; ++half) {
        const v8f& acc = half ? acc1[s] : acc0[s];
        int nt = half ? nt1 : nt0;
        const float4* yp = (const float4*)(yfrag + ((nt * 4 + s) * 32 + lane) * 8);
        float4 y0 = yp[0], y1 = yp[1];           // two ds_load_b128
        float yv[8] = {y0.x, y0.y, y0.z, y0.w, y1.x, y1.y, y1.z, y1.w};
        int col = nt * 16 + nl;
        #pragma unroll
        for (int v = 0; v < 8; ++v) {
          int m = s * 16 + v + hi * 8;
          float a = soft_thr(acc[v] + yv[v], thr[v]);
          nxt[m * 512 + col] = (_Float16)a;
          if (LAST) out_alpha[(size_t)(rowBase + m) * 512 + col] = a;
        }
      }
    }
  }
}

__global__ void ista_kernel(const float* __restrict__ x,
                            const _Float16* __restrict__ wsh,
                            const float* __restrict__ lthr,
                            const float* __restrict__ cptr,
                            float* __restrict__ out_xpred,
                            float* __restrict__ out_alpha) {
  extern __shared__ char smem[];
  int tid  = threadIdx.x;
  int lane = tid & 31;
  int wave = tid >> 5;          // 0..7
  int hi   = lane >> 4;
  int nl   = lane & 15;
  int ntBase = wave * 4;        // 4 n-tiles (64 cols) per wave

  _Float16* alphaA = (_Float16*)(smem + ALPHA_A_OFF);  // 64 x 512 f16
  _Float16* alphaB = (_Float16*)(smem + ALPHA_B_OFF);  // 64 x 512 f16
  float*    yfrag  = (float*)(smem + YFRAG_OFF);       // y/c [nt][strip][lane][v]
  float*    sthr   = (float*)(smem + STHR_OFF);        // 64 thresholds

  const _Float16* Spack  = wsh + SPACK_OFF;
  const _Float16* Dpack  = wsh + DPACK_OFF;
  const _Float16* Dtpack = wsh + DTPACK_OFF;

  int rowBase = blockIdx.x * BLOCK_M;
  float inv_c = 1.0f / cptr[0];

  // stage x tile (64x64) as f16 into alphaB; thresholds into LDS
  _Float16* xstage = alphaB;
  #pragma unroll
  for (int i = 0; i < 16; ++i) {
    int flat = i * THREADS + tid;                // 0..4095
    int r = flat >> 6, q = flat & 63;
    xstage[flat] = (_Float16)x[(rowBase + r) * 64 + q];
  }
  if (tid < BLOCK_M) sthr[tid] = lthr[rowBase + tid];
  __syncthreads();

  // y = x @ Dict ; alpha0 = soft(y, thr) ; yfrag = y/c (fragment-major)
  for (int np = 0; np < 2; ++np) {
    int nt0 = ntBase + np * 2, nt1 = nt0 + 1;
    v8f acc0[4] = {}, acc1[4] = {};
    #pragma unroll
    for (int kc = 0; kc < 2; ++kc) {
      v16h b0 = load_b_frag(Dpack + (kc * 32 + nt0) * 512, lane);
      v16h b1 = load_b_frag(Dpack + (kc * 32 + nt1) * 512, lane);
      #pragma unroll
      for (int s = 0; s < 4; ++s) {
        v16h a = load_a_frag(xstage + s * 16 * 64, 64, kc, lane);
        acc0[s] = __builtin_amdgcn_wmma_f32_16x16x32_f16(false, a, false, b0,
                                                         (short)0, acc0[s],
                                                         false, false);
        acc1[s] = __builtin_amdgcn_wmma_f32_16x16x32_f16(false, a, false, b1,
                                                         (short)0, acc1[s],
                                                         false, false);
      }
    }
    #pragma unroll
    for (int s = 0; s < 4; ++s) {
      const float4* tp = (const float4*)(sthr + s * 16 + hi * 8);
      float4 t0 = tp[0], t1 = tp[1];
      float thr[8] = {t0.x, t0.y, t0.z, t0.w, t1.x, t1.y, t1.z, t1.w};
      #pragma unroll
      for (int half = 0; half < 2; ++half) {
        const v8f& acc = half ? acc1[s] : acc0[s];
        int nt = half ? nt1 : nt0;
        float4* yp = (float4*)(yfrag + ((nt * 4 + s) * 32 + lane) * 8);
        yp[0] = make_float4(acc[0] * inv_c, acc[1] * inv_c,
                            acc[2] * inv_c, acc[3] * inv_c);
        yp[1] = make_float4(acc[4] * inv_c, acc[5] * inv_c,
                            acc[6] * inv_c, acc[7] * inv_c);
        int col = nt * 16 + nl;
        #pragma unroll
        for (int v = 0; v < 8; ++v) {
          int m = s * 16 + v + hi * 8;
          alphaA[m * 512 + col] = (_Float16)soft_thr(acc[v], thr[v]);
        }
      }
    }
  }
  __syncthreads();

  // 15 interior ISTA iterations + 1 peeled final (writes d_out alpha)
  _Float16* cur = alphaA;
  _Float16* nxt = alphaB;
  for (int t = 0; t < T_ITER - 1; ++t) {
    ista_iter<false>(cur, nxt, Spack, yfrag, sthr, out_alpha,
                     rowBase, lane, hi, ntBase);
    _Float16* tmp = cur; cur = nxt; nxt = tmp;
    __syncthreads();
  }
  ista_iter<true>(cur, nxt, Spack, yfrag, sthr, out_alpha,
                  rowBase, lane, hi, ntBase);
  __syncthreads();
  cur = nxt;                                     // final alpha (64 x 512)

  // x_pred = alpha @ Dict^T : 16 output tiles (4 strips x 4 n-tiles),
  // 2 per wave (two strips, one n-tile) -> two independent WMMA chains.
  {
    int nt = wave & 3;
    int s0 = (wave >> 2) * 2, s1 = s0 + 1;
    v8f acc0 = {}, acc1 = {};
    #pragma unroll 4
    for (int kc = 0; kc < 16; ++kc) {
      v16h b  = load_b_frag(Dtpack + (kc * 4 + nt) * 512, lane);
      v16h a0 = load_a_frag(cur + s0 * 16 * 512, 512, kc, lane);
      v16h a1 = load_a_frag(cur + s1 * 16 * 512, 512, kc, lane);
      acc0 = __builtin_amdgcn_wmma_f32_16x16x32_f16(false, a0, false, b,
                                                    (short)0, acc0, false, false);
      acc1 = __builtin_amdgcn_wmma_f32_16x16x32_f16(false, a1, false, b,
                                                    (short)0, acc1, false, false);
    }
    int col = nt * 16 + nl;
    #pragma unroll
    for (int v = 0; v < 8; ++v) {
      int m0 = s0 * 16 + v + hi * 8;
      int m1 = s1 * 16 + v + hi * 8;
      out_xpred[(size_t)(rowBase + m0) * 64 + col] = acc0[v];
      out_xpred[(size_t)(rowBase + m1) * 64 + col] = acc1[v];
    }
  }
}

// ================= host launch =============================================
extern "C" void kernel_launch(void* const* d_in, const int* in_sizes, int n_in,
                              void* d_out, int out_size, void* d_ws, size_t ws_size,
                              hipStream_t stream) {
  (void)in_sizes; (void)n_in; (void)out_size; (void)ws_size;
  const float* x    = (const float*)d_in[0];
  // d_in[1] = N (int, unused by the math)
  const float* W1   = (const float*)d_in[2];
  const float* b1   = (const float*)d_in[3];
  const float* W2   = (const float*)d_in[4];
  const float* b2   = (const float*)d_in[5];
  const float* W3   = (const float*)d_in[6];
  const float* b3   = (const float*)d_in[7];
  const float* W4   = (const float*)d_in[8];
  const float* b4   = (const float*)d_in[9];
  const float* Dict = (const float*)d_in[10];
  const float* c    = (const float*)d_in[11];

  float* out       = (float*)d_out;
  float* out_xpred = out;                                           // B x 64
  float* out_alpha = out + (size_t)B_ROWS * Q_DIM;                  // B x 512
  float* out_lam   = out + (size_t)B_ROWS * Q_DIM
                         + (size_t)B_ROWS * L_DIM;                  // B x 1

  _Float16* wsh  = (_Float16*)d_ws;
  float*    lthr = (float*)((char*)d_ws + LTHR_BYTE_OFF);

  pack_kernel<<<(WS_HALVES + 255) / 256, 256, 0, stream>>>(Dict, c, wsh);
  mlp_kernel<<<B_ROWS / 128, 128, 140416, stream>>>(x, W1, b1, W2, b2, W3, b3,
                                                    W4, b4, c, out_lam, lthr);
  ista_kernel<<<B_ROWS / BLOCK_M, THREADS, LDS_BYTES, stream>>>(
      x, wsh, lthr, c, out_xpred, out_alpha);
}
